// GNNLayers_13649406067041
// MI455X (gfx1250) — matrix operand order repaired and checked by reading, not compiled, not run on hardware
//
#include <hip/hip_runtime.h>

// ---------------------------------------------------------------------------
// EGNN layer for MI455X (gfx1250, wave32, WMMA f32_16x16x32_f16)
// Weights staged to LDS via CDNA5 async copy (global_load_async_to_lds_b128),
// B-fragments served from LDS; A-fragments gathered from global.
// N_IN=11, D=64, DE=4, N_NODES=50000, N_EDGES=800000
// ---------------------------------------------------------------------------

#define N_NODES 50000
#define N_EDGES 800000
#define D 64
#define EPSBN 1e-5f

typedef _Float16 half8_t  __attribute__((ext_vector_type(8)));
typedef _Float16 half16_t __attribute__((ext_vector_type(16)));
typedef float    floatx8  __attribute__((ext_vector_type(8)));

// ---------------- workspace layout (byte offsets, 64B-aligned regions) ----
static constexpr size_t WS_STATS    = 0;                         // 5*128 f32
static constexpr size_t WS_MSG_AGGR = 2560;                      // 50000*64 f32
static constexpr size_t WS_POS_SUM  = WS_MSG_AGGR + 12800000;    // 50000*3 f32
static constexpr size_t WS_CNT      = WS_POS_SUM + 600000;       // 50000 f32
static constexpr size_t WS_ZERO_END = WS_CNT + 200000;           // 13,602,560
static constexpr size_t WS_COEF     = WS_ZERO_END;               // 5*128 f32
static constexpr size_t WS_HF16     = WS_COEF + 2560;            // 50000*64 f16
static constexpr size_t WS_POSDIFF  = WS_HF16 + 6400000;         // 800000*3 f32
static constexpr size_t WS_ZA       = WS_POSDIFF + 9600000;      // 800000*64 f16
static constexpr size_t WS_ZB       = WS_ZA + 102400000;         // 800000*64 f16
static constexpr size_t WS_Z4       = WS_ZB + 102400000;         // 50000*64 f16
static constexpr size_t WS_Z5       = WS_Z4 + 6400000;           // 50000*64 f16
static constexpr size_t WS_W16      = WS_Z5 + 6400000;           // f16 weights
static constexpr size_t WO_MSGW1 = 0;                 // 133*64
static constexpr size_t WO_MSGW2 = 17024;             // 64*64
static constexpr size_t WO_POSW1 = WO_MSGW2 + 8192;
static constexpr size_t WO_UPDW1 = WO_POSW1 + 8192;   // 128*64
static constexpr size_t WO_UPDW2 = WO_UPDW1 + 16384;

static constexpr int ZERO_FLOATS = (int)(WS_ZERO_END / 4);       // 3,400,640

// ---------------- CDNA5 async copy to LDS ----------------------------------
// global_load_async_to_lds_b128: VDST = LDS byte offset (per lane), VADDR =
// 64-bit global address; tracked with ASYNCcnt (CDNA5 ISA)
__device__ __forceinline__ void async_copy_b128(unsigned int lds_off,
                                                const _Float16* gptr) {
    asm volatile("global_load_async_to_lds_b128 %0, %1, off"
                 :: "v"(lds_off), "v"(gptr)
                 : "memory");
}
__device__ __forceinline__ void async_wait0() {
    asm volatile("s_wait_asynccnt 0x0" ::: "memory");
}

// stage `halves` f16 elements of weights into LDS (block-cooperative).
// Caller must __syncthreads() afterwards for cross-wave visibility.
__device__ __forceinline__ void stage_weights(const _Float16* __restrict__ g,
                                              _Float16* lds, int halves) {
    unsigned int base = (unsigned int)(uintptr_t)lds;   // LDS aperture: low 32b = offset
    int chunks = halves >> 3;                           // 16B per chunk
    for (int c = threadIdx.x; c < chunks; c += blockDim.x)
        async_copy_b128(base + (unsigned int)c * 16u, g + c * 8);
    async_wait0();                                      // own-wave ASYNCcnt == 0
}

// ---------------- WMMA helpers ---------------------------------------------
__device__ __forceinline__ floatx8 wmma_f16(half16_t a, half16_t b, floatx8 c) {
    return __builtin_amdgcn_wmma_f32_16x16x32_f16(
        /*neg_a=*/false, a, /*neg_b=*/false, b,
        /*c_mod=*/(short)0, c, /*reuse_a=*/false, /*reuse_b=*/false);
}

__device__ __forceinline__ half16_t combine8(half8_t a, half8_t b) {
    half16_t r;
#pragma unroll
    for (int i = 0; i < 8; ++i) { r[i] = a[i]; r[8 + i] = b[i]; }
    return r;
}

// A-fragment (16x32 f16): lane&15 = row M; lanes>=16 take K+8 halves.
// Per lane: halves 0..7 = K[base..base+8), halves 8..15 = K[base+16..base+24)
__device__ __forceinline__ half16_t make_afrag(const _Float16* __restrict__ rowptr,
                                               int koff, int base) {
    half8_t c0 = *(const half8_t*)(rowptr + koff + base);
    half8_t c1 = *(const half8_t*)(rowptr + koff + base + 16);
    return combine8(c0, c1);
}

__device__ __forceinline__ half16_t make_afrag_f32(const float* __restrict__ rowptr,
                                                   int koff, int base) {
    half16_t a;
#pragma unroll
    for (int j = 0; j < 2; ++j) {
        int kb = koff + base + j * 16;
#pragma unroll
        for (int i = 0; i < 8; ++i) a[j * 8 + i] = (_Float16)rowptr[kb + i];
    }
    return a;
}

// BN(scale,shift)+ReLU applied while building the A-fragment
__device__ __forceinline__ half16_t make_afrag_bnrelu(const _Float16* __restrict__ rowptr,
                                                      int koff, int base,
                                                      const float* __restrict__ sc,
                                                      const float* __restrict__ sh) {
    half16_t a;
#pragma unroll
    for (int j = 0; j < 2; ++j) {
        int kb = koff + base + j * 16;
#pragma unroll
        for (int i = 0; i < 8; ++i) {
            float v = (float)rowptr[kb + i] * sc[kb + i] + sh[kb + i];
            v = v > 0.f ? v : 0.f;
            a[j * 8 + i] = (_Float16)v;
        }
    }
    return a;
}

// B-fragment (32x16 f16): lane&15 = N column; lanes>=16 take K+8.
// Per lane: halves 0..7 = K[base..base+8), halves 8..15 = K[base+16..base+24)
__device__ __forceinline__ half16_t load_bfrag(const _Float16* W,
                                               int krow0, int nt, int lane) {
    const _Float16* p = W + (size_t)(krow0 + ((lane & 16) ? 8 : 0)) * 64 + nt * 16 + (lane & 15);
    half16_t r;
#pragma unroll
    for (int i = 0; i < 8; ++i) { r[i] = p[i * 64]; r[8 + i] = p[(16 + i) * 64]; }
    return r;
}

// ---------------- small utility kernels ------------------------------------
__global__ void zero_kernel(float* __restrict__ p, int n) {
    int i = blockIdx.x * blockDim.x + threadIdx.x;
    if (i < n) p[i] = 0.f;
}

__global__ void cvt_f16_kernel(const float* __restrict__ s, _Float16* __restrict__ d, int n) {
    int i = blockIdx.x * blockDim.x + threadIdx.x;
    if (i < n) d[i] = (_Float16)s[i];
}

// h = x @ lin_W + lin_b  (50000 x 11) @ (11 x 64), stored f16
__global__ void node_proj_kernel(const float* __restrict__ x,
                                 const float* __restrict__ W,
                                 const float* __restrict__ b,
                                 _Float16* __restrict__ hf) {
    int i = blockIdx.x * blockDim.x + threadIdx.x;
    if (i >= N_NODES * D) return;
    int n = i >> 6, c = i & 63;
    float acc = b[c];
#pragma unroll
    for (int k = 0; k < 11; ++k) acc += x[n * 11 + k] * W[k * 64 + c];
    hf[i] = (_Float16)acc;
}

// mean/var -> per-column scale/shift
__global__ void bn_finalize_kernel(const float* __restrict__ stats,
                                   const float* __restrict__ gamma,
                                   const float* __restrict__ beta,
                                   float* __restrict__ coef, float N) {
    int t = threadIdx.x;
    if (t >= 64) return;
    float mean = stats[t] / N;
    float var = stats[64 + t] / N - mean * mean;
    float sc = gamma[t] * rsqrtf(var + EPSBN);
    coef[t] = sc;
    coef[64 + t] = beta[t] - sc * mean;
}

// ---------------- edge pass 1: z1 = [h_i,h_j,radial,ea] @ W1 + b1 ----------
__global__ void edge_msg1_kernel(const float* __restrict__ pos,
                                 const int* __restrict__ src,
                                 const int* __restrict__ dst,
                                 const float* __restrict__ ea,
                                 const _Float16* __restrict__ hf,
                                 const _Float16* __restrict__ W1f,  // 133x64
                                 const float* __restrict__ b1,
                                 _Float16* __restrict__ z1,
                                 float* __restrict__ pos_diff,
                                 float* __restrict__ cnt,
                                 float* __restrict__ stats) {
    __shared__ _Float16 sW[133 * 64];
    __shared__ float s_rad[8][16];
    __shared__ float s_ea[8][16][4];
    __shared__ float s_sum[64], s_sumsq[64];
    int t = threadIdx.x, w = t >> 5, lane = t & 31, l15 = lane & 15;

    stage_weights(W1f, sW, 133 * 64);          // async -> LDS, own-wave waited
    if (t < 64) { s_sum[t] = 0.f; s_sumsq[t] = 0.f; }

    int tile = blockIdx.x * 8 + w;             // exact: 6250*8 = 50000 tiles
    int e0 = tile * 16;
    int e = e0 + l15;
    int sj = src[e], di = dst[e];
    if (lane < 16) {
        float dx = pos[di * 3 + 0] - pos[sj * 3 + 0];
        float dy = pos[di * 3 + 1] - pos[sj * 3 + 1];
        float dz = pos[di * 3 + 2] - pos[sj * 3 + 2];
        pos_diff[e * 3 + 0] = dx; pos_diff[e * 3 + 1] = dy; pos_diff[e * 3 + 2] = dz;
        s_rad[w][l15] = dx * dx + dy * dy + dz * dz;
#pragma unroll
        for (int q = 0; q < 4; ++q) s_ea[w][l15][q] = ea[e * 4 + q];
        atomicAdd(&cnt[di], 1.0f);
    }
    __syncthreads();                           // weights + per-tile scalars visible

    const _Float16* pi = hf + (size_t)di * 64;
    const _Float16* pj = hf + (size_t)sj * 64;
    int base = (lane & 16) ? 8 : 0;
    half16_t ai0 = make_afrag(pi, 0, base),  ai1 = make_afrag(pi, 32, base);
    half16_t aj0 = make_afrag(pj, 0, base),  aj1 = make_afrag(pj, 32, base);
    int mo = (lane >> 4) * 8;

#pragma unroll
    for (int nt = 0; nt < 4; ++nt) {
        floatx8 acc = {};
        acc = wmma_f16(ai0, load_bfrag(sW, 0,  nt, lane), acc);
        acc = wmma_f16(ai1, load_bfrag(sW, 32, nt, lane), acc);
        acc = wmma_f16(aj0, load_bfrag(sW, 64, nt, lane), acc);
        acc = wmma_f16(aj1, load_bfrag(sW, 96, nt, lane), acc);
        int col = l15 + nt * 16;
        float wr = (float)sW[128 * 64 + col];
        float w0 = (float)sW[129 * 64 + col], w1 = (float)sW[130 * 64 + col];
        float w2 = (float)sW[131 * 64 + col], w3 = (float)sW[132 * 64 + col];
        float bb = b1[col], ssum = 0.f, ssq = 0.f;
#pragma unroll
        for (int r = 0; r < 8; ++r) {
            int m = mo + r;
            float z = acc[r] + bb + s_rad[w][m] * wr +
                      s_ea[w][m][0] * w0 + s_ea[w][m][1] * w1 +
                      s_ea[w][m][2] * w2 + s_ea[w][m][3] * w3;
            z1[(size_t)(e0 + m) * 64 + col] = (_Float16)z;
            ssum += z; ssq += z * z;
        }
        atomicAdd(&s_sum[col], ssum);
        atomicAdd(&s_sumsq[col], ssq);
    }
    __syncthreads();
    if (t < 64) { atomicAdd(&stats[t], s_sum[t]); atomicAdd(&stats[64 + t], s_sumsq[t]); }
}

// ---------------- generic pass: zout = relu(bn(zin)) @ W + b ---------------
__global__ void bnrelu_gemm_kernel(const _Float16* __restrict__ zin,
                                   const float* __restrict__ coef,
                                   const _Float16* __restrict__ W,   // 64x64
                                   const float* __restrict__ bias,
                                   _Float16* __restrict__ zout,
                                   float* __restrict__ stats, int ntiles) {
    __shared__ _Float16 sW[64 * 64];
    __shared__ float s_sum[64], s_sumsq[64];
    int t = threadIdx.x, w = t >> 5, lane = t & 31, l15 = lane & 15;
    stage_weights(W, sW, 64 * 64);
    if (t < 64) { s_sum[t] = 0.f; s_sumsq[t] = 0.f; }
    __syncthreads();
    int tile = blockIdx.x * 8 + w;
    if (tile < ntiles) {                      // wave-uniform predicate
        int e0 = tile * 16;
        int base = (lane & 16) ? 8 : 0;
        const _Float16* rp = zin + (size_t)(e0 + l15) * 64;
        half16_t a0 = make_afrag_bnrelu(rp, 0,  base, coef, coef + 64);
        half16_t a1 = make_afrag_bnrelu(rp, 32, base, coef, coef + 64);
        int mo = (lane >> 4) * 8;
#pragma unroll
        for (int nt = 0; nt < 4; ++nt) {
            floatx8 acc = {};
            acc = wmma_f16(a0, load_bfrag(sW, 0,  nt, lane), acc);
            acc = wmma_f16(a1, load_bfrag(sW, 32, nt, lane), acc);
            int col = l15 + nt * 16;
            float bb = bias[col], ssum = 0.f, ssq = 0.f;
#pragma unroll
            for (int r = 0; r < 8; ++r) {
                float z = acc[r] + bb;
                zout[(size_t)(e0 + mo + r) * 64 + col] = (_Float16)z;
                ssum += z; ssq += z * z;
            }
            atomicAdd(&s_sum[col], ssum);
            atomicAdd(&s_sumsq[col], ssq);
        }
    }
    __syncthreads();
    if (t < 64) { atomicAdd(&stats[t], s_sum[t]); atomicAdd(&stats[64 + t], s_sumsq[t]); }
}

// ---- edge pass 3: msg = relu(bn2(z2)); scatter msg; z3 = msg @ posW1 + b --
__global__ void edge_msg3_kernel(const _Float16* __restrict__ z2,
                                 const float* __restrict__ coef2,
                                 const int* __restrict__ dst,
                                 const _Float16* __restrict__ P1f,  // 64x64
                                 const float* __restrict__ pb1,
                                 float* __restrict__ msg_aggr,
                                 _Float16* __restrict__ z3,
                                 float* __restrict__ stats) {
    __shared__ _Float16 sW[64 * 64];
    __shared__ float s_sum[64], s_sumsq[64];
    int t = threadIdx.x, w = t >> 5, lane = t & 31, l15 = lane & 15;
    stage_weights(P1f, sW, 64 * 64);
    if (t < 64) { s_sum[t] = 0.f; s_sumsq[t] = 0.f; }
    __syncthreads();
    int tile = blockIdx.x * 8 + w;            // exact 50000 tiles
    int e0 = tile * 16;
    int e = e0 + l15;
    int d = dst[e];
    int base = (lane & 16) ? 8 : 0;
    const _Float16* rp = z2 + (size_t)e * 64;
    const float* sc = coef2; const float* sh = coef2 + 64;

    half16_t m0, m1;
#pragma unroll
    for (int kk = 0; kk < 2; ++kk) {
#pragma unroll
        for (int j = 0; j < 2; ++j) {
#pragma unroll
            for (int i = 0; i < 8; ++i) {
                int k = kk * 32 + base + j * 16 + i;
                float v = (float)rp[k] * sc[k] + sh[k];
                v = v > 0.f ? v : 0.f;
                atomicAdd(&msg_aggr[(size_t)d * 64 + k], v);  // segment_sum(msg)
                if (kk == 0) m0[j * 8 + i] = (_Float16)v;
                else         m1[j * 8 + i] = (_Float16)v;
            }
        }
    }
    int mo = (lane >> 4) * 8;
#pragma unroll
    for (int nt = 0; nt < 4; ++nt) {
        floatx8 acc = {};
        acc = wmma_f16(m0, load_bfrag(sW, 0,  nt, lane), acc);
        acc = wmma_f16(m1, load_bfrag(sW, 32, nt, lane), acc);
        int col = l15 + nt * 16;
        float bb = pb1[col], ssum = 0.f, ssq = 0.f;
#pragma unroll
        for (int r = 0; r < 8; ++r) {
            float z = acc[r] + bb;
            z3[(size_t)(e0 + mo + r) * 64 + col] = (_Float16)z;
            ssum += z; ssq += z * z;
        }
        atomicAdd(&s_sum[col], ssum);
        atomicAdd(&s_sumsq[col], ssq);
    }
    __syncthreads();
    if (t < 64) { atomicAdd(&stats[t], s_sum[t]); atomicAdd(&stats[64 + t], s_sumsq[t]); }
}

// ---- pos branch: s = relu(bn3(z3)) . posW2 + b; scatter pos_diff*s --------
__global__ void edge_pos_kernel(const _Float16* __restrict__ z3,
                                const float* __restrict__ coef3,
                                const float* __restrict__ pW2,   // 64
                                const float* __restrict__ pb2,   // 1
                                const float* __restrict__ pos_diff,
                                const int* __restrict__ dst,
                                float* __restrict__ pos_sum) {
    int e = blockIdx.x * blockDim.x + threadIdx.x;
    if (e >= N_EDGES) return;
    const _Float16* rp = z3 + (size_t)e * 64;
    float s = pb2[0];
#pragma unroll
    for (int k = 0; k < 64; ++k) {
        float q = (float)rp[k] * coef3[k] + coef3[64 + k];
        q = q > 0.f ? q : 0.f;
        s += q * pW2[k];
    }
    int d = dst[e];
#pragma unroll
    for (int j = 0; j < 3; ++j)
        atomicAdd(&pos_sum[d * 3 + j], pos_diff[e * 3 + j] * s);
}

// ---- node pass 1: z4 = [h, msg_aggr] @ updW1 + b --------------------------
__global__ void node_upd1_kernel(const _Float16* __restrict__ hf,
                                 const float* __restrict__ msg_aggr,
                                 const _Float16* __restrict__ U1f,  // 128x64
                                 const float* __restrict__ b,
                                 _Float16* __restrict__ z4,
                                 float* __restrict__ stats, int ntiles) {
    __shared__ _Float16 sW[128 * 64];
    __shared__ float s_sum[64], s_sumsq[64];
    int t = threadIdx.x, w = t >> 5, lane = t & 31, l15 = lane & 15;
    stage_weights(U1f, sW, 128 * 64);
    if (t < 64) { s_sum[t] = 0.f; s_sumsq[t] = 0.f; }
    __syncthreads();
    int tile = blockIdx.x * 8 + w;
    if (tile < ntiles) {
        int n0 = tile * 16;
        int base = (lane & 16) ? 8 : 0;
        const _Float16* hp = hf + (size_t)(n0 + l15) * 64;
        const float* mp = msg_aggr + (size_t)(n0 + l15) * 64;
        half16_t h0 = make_afrag(hp, 0, base),      h1 = make_afrag(hp, 32, base);
        half16_t g0 = make_afrag_f32(mp, 0, base),  g1 = make_afrag_f32(mp, 32, base);
        int mo = (lane >> 4) * 8;
#pragma unroll
        for (int nt = 0; nt < 4; ++nt) {
            floatx8 acc = {};
            acc = wmma_f16(h0, load_bfrag(sW, 0,  nt, lane), acc);
            acc = wmma_f16(h1, load_bfrag(sW, 32, nt, lane), acc);
            acc = wmma_f16(g0, load_bfrag(sW, 64, nt, lane), acc);
            acc = wmma_f16(g1, load_bfrag(sW, 96, nt, lane), acc);
            int col = l15 + nt * 16;
            float bb = b[col], ssum = 0.f, ssq = 0.f;
#pragma unroll
            for (int r = 0; r < 8; ++r) {
                float z = acc[r] + bb;
                z4[(size_t)(n0 + mo + r) * 64 + col] = (_Float16)z;
                ssum += z; ssq += z * z;
            }
            atomicAdd(&s_sum[col], ssum);
            atomicAdd(&s_sumsq[col], ssq);
        }
    }
    __syncthreads();
    if (t < 64) { atomicAdd(&stats[t], s_sum[t]); atomicAdd(&stats[64 + t], s_sumsq[t]); }
}

// ---- finals ---------------------------------------------------------------
__global__ void final_h_kernel(const _Float16* __restrict__ z5,
                               const float* __restrict__ coef5,
                               float* __restrict__ out) {
    int i = blockIdx.x * blockDim.x + threadIdx.x;
    if (i >= N_NODES * D) return;
    int c = i & 63;
    float v = (float)z5[i] * coef5[c] + coef5[64 + c];
    out[i] = v > 0.f ? v : 0.f;
}

__global__ void final_pos_kernel(const float* __restrict__ pos,
                                 const float* __restrict__ pos_sum,
                                 const float* __restrict__ cnt,
                                 float* __restrict__ out) {
    int i = blockIdx.x * blockDim.x + threadIdx.x;
    if (i >= N_NODES * 3) return;
    int n = i / 3;
    float c = cnt[n]; c = c < 1.f ? 1.f : c;
    out[i] = pos[i] + pos_sum[i] / c;
}

// ---------------------------------------------------------------------------
extern "C" void kernel_launch(void* const* d_in, const int* in_sizes, int n_in,
                              void* d_out, int out_size, void* d_ws, size_t ws_size,
                              hipStream_t stream) {
    const float* x         = (const float*)d_in[0];
    const float* pos       = (const float*)d_in[1];
    const int*   ei        = (const int*)d_in[2];
    const float* edge_attr = (const float*)d_in[3];
    const float* lin_W = (const float*)d_in[4];  const float* lin_b = (const float*)d_in[5];
    const float* msgW1 = (const float*)d_in[6];  const float* msgb1 = (const float*)d_in[7];
    const float* msgg1 = (const float*)d_in[8];  const float* msgbe1= (const float*)d_in[9];
    const float* msgW2 = (const float*)d_in[10]; const float* msgb2 = (const float*)d_in[11];
    const float* msgg2 = (const float*)d_in[12]; const float* msgbe2= (const float*)d_in[13];
    const float* posW1 = (const float*)d_in[14]; const float* posb1 = (const float*)d_in[15];
    const float* posg1 = (const float*)d_in[16]; const float* posbe1= (const float*)d_in[17];
    const float* posW2 = (const float*)d_in[18]; const float* posb2 = (const float*)d_in[19];
    const float* updW1 = (const float*)d_in[20]; const float* updb1 = (const float*)d_in[21];
    const float* updg1 = (const float*)d_in[22]; const float* updbe1= (const float*)d_in[23];
    const float* updW2 = (const float*)d_in[24]; const float* updb2 = (const float*)d_in[25];
    const float* updg2 = (const float*)d_in[26]; const float* updbe2= (const float*)d_in[27];

    const int* src = ei;
    const int* dst = ei + N_EDGES;

    char* w = (char*)d_ws;
    float*     stats    = (float*)(w + WS_STATS);     // 5 sets of 128
    float*     msg_aggr = (float*)(w + WS_MSG_AGGR);
    float*     pos_sum  = (float*)(w + WS_POS_SUM);
    float*     cnt      = (float*)(w + WS_CNT);
    float*     coef     = (float*)(w + WS_COEF);      // 5 sets of 128
    _Float16*  hf16     = (_Float16*)(w + WS_HF16);
    float*     posdiff  = (float*)(w + WS_POSDIFF);
    _Float16*  zA       = (_Float16*)(w + WS_ZA);
    _Float16*  zB       = (_Float16*)(w + WS_ZB);
    _Float16*  z4       = (_Float16*)(w + WS_Z4);
    _Float16*  z5       = (_Float16*)(w + WS_Z5);
    _Float16*  msgW1f   = (_Float16*)(w + WS_W16 + WO_MSGW1);
    _Float16*  msgW2f   = (_Float16*)(w + WS_W16 + WO_MSGW2);
    _Float16*  posW1f   = (_Float16*)(w + WS_W16 + WO_POSW1);
    _Float16*  updW1f   = (_Float16*)(w + WS_W16 + WO_UPDW1);
    _Float16*  updW2f   = (_Float16*)(w + WS_W16 + WO_UPDW2);

    // 0) zero accumulators (stats + msg_aggr + pos_sum + cnt)
    zero_kernel<<<(ZERO_FLOATS + 255) / 256, 256, 0, stream>>>((float*)w, ZERO_FLOATS);

    // 1) f16 weight copies
    cvt_f16_kernel<<<(133 * 64 + 255) / 256, 256, 0, stream>>>(msgW1, msgW1f, 133 * 64);
    cvt_f16_kernel<<<(64 * 64 + 255) / 256, 256, 0, stream>>>(msgW2, msgW2f, 64 * 64);
    cvt_f16_kernel<<<(64 * 64 + 255) / 256, 256, 0, stream>>>(posW1, posW1f, 64 * 64);
    cvt_f16_kernel<<<(128 * 64 + 255) / 256, 256, 0, stream>>>(updW1, updW1f, 128 * 64);
    cvt_f16_kernel<<<(64 * 64 + 255) / 256, 256, 0, stream>>>(updW2, updW2f, 64 * 64);

    // 2) node projection
    node_proj_kernel<<<(N_NODES * D) / 256, 256, 0, stream>>>(x, lin_W, lin_b, hf16);

    // 3) edge linear 1 (+radial/edge_attr tail) -> z1, stats0
    edge_msg1_kernel<<<N_EDGES / 128, 256, 0, stream>>>(pos, src, dst, edge_attr, hf16,
                                                        msgW1f, msgb1, zA, posdiff, cnt,
                                                        stats + 0);
    bn_finalize_kernel<<<1, 64, 0, stream>>>(stats + 0, msgg1, msgbe1, coef + 0, (float)N_EDGES);

    // 4) z2 = relu(bn1(z1)) @ msgW2, stats1
    bnrelu_gemm_kernel<<<N_EDGES / 128, 256, 0, stream>>>(zA, coef + 0, msgW2f, msgb2,
                                                          zB, stats + 128, N_EDGES / 16);
    bn_finalize_kernel<<<1, 64, 0, stream>>>(stats + 128, msgg2, msgbe2, coef + 128, (float)N_EDGES);

    // 5) msg = relu(bn2(z2)); scatter msg; z3 = msg @ posW1, stats2
    edge_msg3_kernel<<<N_EDGES / 128, 256, 0, stream>>>(zB, coef + 128, dst, posW1f, posb1,
                                                        msg_aggr, zA, stats + 256);
    bn_finalize_kernel<<<1, 64, 0, stream>>>(stats + 256, posg1, posbe1, coef + 256, (float)N_EDGES);

    // 6) position scalar + scatter pos updates
    edge_pos_kernel<<<N_EDGES / 256, 256, 0, stream>>>(zA, coef + 256, posW2, posb2,
                                                       posdiff, dst, pos_sum);

    // 7) node update MLP
    node_upd1_kernel<<<(N_NODES / 16 + 7) / 8, 256, 0, stream>>>(hf16, msg_aggr, updW1f, updb1,
                                                                 z4, stats + 384, N_NODES / 16);
    bn_finalize_kernel<<<1, 64, 0, stream>>>(stats + 384, updg1, updbe1, coef + 384, (float)N_NODES);
    bnrelu_gemm_kernel<<<(N_NODES / 16 + 7) / 8, 256, 0, stream>>>(z4, coef + 384, updW2f, updb2,
                                                                   z5, stats + 512, N_NODES / 16);
    bn_finalize_kernel<<<1, 64, 0, stream>>>(stats + 512, updg2, updbe2, coef + 512, (float)N_NODES);

    // 8) outputs: h_out (50000x64) then pos_out (50000x3), flat f32
    float* out = (float*)d_out;
    final_h_kernel<<<(N_NODES * D) / 256, 256, 0, stream>>>(z5, coef + 512, out);
    final_pos_kernel<<<(N_NODES * 3 + 255) / 256, 256, 0, stream>>>(pos, pos_sum, cnt,
                                                                    out + N_NODES * D);
}
